// EncoderProcessDecoder_13348758356101
// MI455X (gfx1250) — matrix operand order repaired and checked by reading, not compiled, not run on hardware
//
#include <hip/hip_runtime.h>
#include <hip/hip_bf16.h>
#include <cstdint>
#include <cstddef>
#include <cmath>

typedef _Float16 half_t;
typedef __attribute__((ext_vector_type(16))) _Float16 v16h;
typedef __attribute__((ext_vector_type(8)))  _Float16 v8h;
typedef __attribute__((ext_vector_type(8)))  float    v8f;

static const int EMB = 128;

// ======================= WMMA GEMM =======================
// C[M,N] = A[M,Kp] @ W (+bias, optional relu), where Bt[n*Kp+k] = W[k][n] (f16).
// A is f16 [M,Kp] (Kp multiple of 32). N multiple of 128.
// grid=(ceil(M/128), N/128), block=256 (8 waves); each wave computes a 16x128
// strip with 8 f32 accumulators. Software-pipelined: fragments for K-step k+32
// are loaded before the 8 WMMAs of step k issue, hiding L2 latency under the
// matrix pipe. Fragment layouts follow CDNA5 ISA 7.12.2 (16-bit A 16x32, B 32x16).
union Frag { v16h v; v8h h[2]; };

__device__ __forceinline__ void load_frags(const half_t* __restrict__ aRow,
                                           const half_t* __restrict__ bBase,
                                           size_t bStride, int k,
                                           Frag& a, Frag* b)
{
    // A: lane 0-15 halves 0..7 = K k..k+7, halves 8..15 = K k+16..k+23;
    //    lanes 16-31 shifted by 8 (folded into aRow).
    a.h[0] = *(const v8h*)(aRow + k);
    a.h[1] = *(const v8h*)(aRow + k + 16);
#pragma unroll
    for (int nt = 0; nt < 8; ++nt) {
        const half_t* bRow = bBase + (size_t)nt * bStride + k;
        b[nt].h[0] = *(const v8h*)(bRow);      // K k..k+7 (lanes 16-31: +8) of col n
        b[nt].h[1] = *(const v8h*)(bRow + 16); // K k+16..k+23 (lanes 16-31: +8)
    }
}

__launch_bounds__(256)
__global__ void k_gemm_wmma(const half_t* __restrict__ A, const half_t* __restrict__ Bt,
                            const float* __restrict__ bias, float* __restrict__ Cf,
                            half_t* __restrict__ Ch, int M, int N, int Kp, int relu)
{
    const int lane   = threadIdx.x & 31;
    const int wave   = threadIdx.x >> 5;
    const int laneLo = lane & 15;
    const int laneHi = lane >> 4;
    const int m0 = (blockIdx.x * 8 + wave) * 16;
    if (m0 >= M) return;                       // uniform per wave
    const int n0 = blockIdx.y * 128;

    v8f acc[8] = {};

    // Out-of-range A rows only feed D rows that the epilogue masks, so just
    // clamp the pointer (no divergent zero-fill needed; EXEC stays all-ones).
    int mA = m0 + laneLo; if (mA >= M) mA = M - 1;
    const half_t* aRow  = A  + (size_t)mA * Kp + laneHi * 8;
    const half_t* bBase = Bt + (size_t)(n0 + laneLo) * Kp + laneHi * 8;
    const size_t  bStride = (size_t)16 * Kp;

    Frag a0, a1; Frag b0[8], b1[8];
    load_frags(aRow, bBase, bStride, 0, a0, b0);

    int k = 0;
    for (;;) {
        if (k + 32 < Kp) load_frags(aRow, bBase, bStride, k + 32, a1, b1);
        __builtin_prefetch(aRow + k + 256, 0, 1);   // global_prefetch_b8
#pragma unroll
        for (int nt = 0; nt < 8; ++nt)
            acc[nt] = __builtin_amdgcn_wmma_f32_16x16x32_f16(
                false, a0.v, false, b0[nt].v, (short)0, acc[nt], false, false);
        k += 32;
        if (k >= Kp) break;

        if (k + 32 < Kp) load_frags(aRow, bBase, bStride, k + 32, a0, b0);
#pragma unroll
        for (int nt = 0; nt < 8; ++nt)
            acc[nt] = __builtin_amdgcn_wmma_f32_16x16x32_f16(
                false, a1.v, false, b1[nt].v, (short)0, acc[nt], false, false);
        k += 32;
        if (k >= Kp) break;
    }

#pragma unroll
    for (int nt = 0; nt < 8; ++nt) {
        const int n = n0 + nt * 16 + laneLo;
        const float bv = bias ? bias[n] : 0.f;
#pragma unroll
        for (int i = 0; i < 8; ++i) {
            const int m = m0 + laneHi * 8 + i;  // C/D layout: lanes16-31 = rows M+8
            if (m >= M) continue;
            float x = acc[nt][i] + bv;
            if (relu) x = fmaxf(x, 0.f);
            const size_t o = (size_t)m * N + n;
            if (Cf) Cf[o] = x;
            if (Ch) Ch[o] = (half_t)x;
        }
    }
}

// ======================= helper kernels =======================
__global__ void k_prep_weight(const float* __restrict__ W, half_t* __restrict__ Wt,
                              int K, int Kp)
{   // W[K,128] f32 -> Wt[128,Kp] f16 transposed, zero-padded in K
    size_t idx = (size_t)blockIdx.x * 256 + threadIdx.x;
    if (idx >= (size_t)EMB * Kp) return;
    int n = (int)(idx / Kp), k = (int)(idx % Kp);
    Wt[idx] = (k < K) ? (half_t)W[(size_t)k * EMB + n] : (half_t)0.f;
}

__global__ void k_fill_h16(half_t* p, size_t n)
{   size_t i = (size_t)blockIdx.x * 256 + threadIdx.x; if (i < n) p[i] = (half_t)0.f; }

__global__ void k_fill_u32(unsigned* p, size_t n, unsigned v)
{   size_t i = (size_t)blockIdx.x * 256 + threadIdx.x; if (i < n) p[i] = v; }

__global__ void k_zero_f32(float* p, size_t n)
{   size_t i = (size_t)blockIdx.x * 256 + threadIdx.x; if (i < n) p[i] = 0.f; }

__global__ void k_pack_node_in(const float* __restrict__ v, const float* __restrict__ goal,
                               half_t* __restrict__ out, int Nn)
{   // [v, g, (v-g)^2, v-g] -> [Nn,32] f16 (C=7, pad to 32)
    size_t idx = (size_t)blockIdx.x * 256 + threadIdx.x;
    if (idx >= (size_t)Nn * 32) return;
    int i = (int)(idx >> 5), j = (int)(idx & 31);
    float val = 0.f;
    if (j < 7)       val = v[i * 7 + j];
    else if (j < 14) val = goal[j - 7];
    else if (j < 21) { float d = v[i * 7 + (j - 14)] - goal[j - 14]; val = d * d; }
    else if (j < 28) val = v[i * 7 + (j - 21)] - goal[j - 21];
    out[idx] = (half_t)val;
}

__global__ void k_pack_edge_in(const float* __restrict__ v, const int* __restrict__ src,
                               const int* __restrict__ dst, half_t* __restrict__ out, int Ne)
{   // [v[src], v[dst]] -> [Ne,32] f16
    size_t idx = (size_t)blockIdx.x * 256 + threadIdx.x;
    if (idx >= (size_t)Ne * 32) return;
    int e = (int)(idx >> 5), j = (int)(idx & 31);
    float val = 0.f;
    if (j < 7)       val = v[(size_t)src[e] * 7 + j];
    else if (j < 14) val = v[(size_t)dst[e] * 7 + (j - 7)];
    out[idx] = (half_t)val;
}

__global__ void k_pack_pad(const float* __restrict__ x, half_t* __restrict__ out,
                           int M, int K, int Kp)
{   size_t idx = (size_t)blockIdx.x * 256 + threadIdx.x;
    if (idx >= (size_t)M * Kp) return;
    int i = (int)(idx / Kp), j = (int)(idx % Kp);
    out[idx] = (j < K) ? (half_t)x[(size_t)i * K + j] : (half_t)0.f;
}

__global__ void k_transpose_pad(const float* __restrict__ X, half_t* __restrict__ out,
                                int R, int Kp)
{   // X[R,128] f32 -> out[128,Kp] f16 (zero pad cols >= R)
    size_t idx = (size_t)blockIdx.x * 256 + threadIdx.x;
    if (idx >= (size_t)EMB * Kp) return;
    int n = (int)(idx / Kp), k = (int)(idx % Kp);
    out[idx] = (k < R) ? (half_t)X[(size_t)k * EMB + n] : (half_t)0.f;
}

__launch_bounds__(128)
__global__ void k_softmax_att(const float* __restrict__ mq, const float* __restrict__ mk,
                              half_t* __restrict__ sc, float* __restrict__ att0,
                              int M, int ncol, int npad, float invt)
{   // one wave per row; softmax over {self} U obstacle scores; in-place normalize f16
    const int lane = threadIdx.x & 31;
    const int r = blockIdx.x * 4 + (threadIdx.x >> 5);
    if (r >= M) return;
    const float* q  = mq + (size_t)r * EMB;
    const float* kk = mk + (size_t)r * EMB;
    float s = 0.f;
#pragma unroll
    for (int i = 0; i < 4; ++i) s += q[lane + 32 * i] * kk[lane + 32 * i];
#pragma unroll
    for (int d = 16; d >= 1; d >>= 1) s += __shfl_xor(s, d, 32);
    const float self_s = s * invt;
    half_t* row = sc + (size_t)r * npad;
    float mx = self_s;
    for (int c = lane; c < ncol; c += 32) mx = fmaxf(mx, (float)row[c] * invt);
#pragma unroll
    for (int d = 16; d >= 1; d >>= 1) mx = fmaxf(mx, __shfl_xor(mx, d, 32));
    float sum = 0.f;
    for (int c = lane; c < npad; c += 32) {
        float e = 0.f;
        if (c < ncol) { e = __expf((float)row[c] * invt - mx); sum += e; }
        row[c] = (half_t)e;
    }
#pragma unroll
    for (int d = 16; d >= 1; d >>= 1) sum += __shfl_xor(sum, d, 32);
    const float e0  = __expf(self_s - mx);
    const float inv = 1.f / (sum + e0);
    for (int c = lane; c < npad; c += 32) row[c] = (half_t)((float)row[c] * inv);
    if (lane == 0) att0[r] = e0 * inv;
}

__launch_bounds__(128)
__global__ void k_res_ln(const float* __restrict__ a, const float* __restrict__ b,
                         const float* __restrict__ c, const float* __restrict__ s,
                         const float* __restrict__ lnw, const float* __restrict__ lnb,
                         float* __restrict__ of, half_t* __restrict__ oh, int M)
{   // x = (s?s[r]:1)*a + b + (c?c:0);  out = LN(x)*lnw + lnb ; one wave per row
    const int lane = threadIdx.x & 31;
    const int r = blockIdx.x * 4 + (threadIdx.x >> 5);
    if (r >= M) return;
    const size_t base = (size_t)r * EMB;
    const float sc = s ? s[r] : 1.f;
    float x[4]; float mean = 0.f;
#pragma unroll
    for (int i = 0; i < 4; ++i) {
        int j = lane + 32 * i;
        float v = sc * a[base + j] + b[base + j] + (c ? c[base + j] : 0.f);
        x[i] = v; mean += v;
    }
#pragma unroll
    for (int d = 16; d >= 1; d >>= 1) mean += __shfl_xor(mean, d, 32);
    mean *= (1.f / 128.f);
    float var = 0.f;
#pragma unroll
    for (int i = 0; i < 4; ++i) { float dv = x[i] - mean; var += dv * dv; }
#pragma unroll
    for (int d = 16; d >= 1; d >>= 1) var += __shfl_xor(var, d, 32);
    var *= (1.f / 128.f);
    const float rstd = rsqrtf(var + 1e-6f);
#pragma unroll
    for (int i = 0; i < 4; ++i) {
        int j = lane + 32 * i;
        float v = (x[i] - mean) * rstd * lnw[j] + lnb[j];
        if (of) of[base + j] = v;
        if (oh) oh[base + j] = (half_t)v;
    }
}

__global__ void k_argmin_goal(const float* __restrict__ v, const float* __restrict__ goal,
                              int* __restrict__ gidx, int Nn)
{
    __shared__ float sd[256]; __shared__ int si[256];
    float best = 3.4e38f; int bi = 0;
    for (int i = threadIdx.x; i < Nn; i += 256) {
        float d = 0.f;
        for (int c = 0; c < 7; ++c) { float t = v[(size_t)i * 7 + c] - goal[c]; d += t * t; }
        if (d < best) { best = d; bi = i; }
    }
    sd[threadIdx.x] = best; si[threadIdx.x] = bi;
    __syncthreads();
    for (int s = 128; s > 0; s >>= 1) {
        if (threadIdx.x < s) {
            if (sd[threadIdx.x + s] < sd[threadIdx.x] ||
                (sd[threadIdx.x + s] == sd[threadIdx.x] && si[threadIdx.x + s] < si[threadIdx.x])) {
                sd[threadIdx.x] = sd[threadIdx.x + s]; si[threadIdx.x] = si[threadIdx.x + s];
            }
        }
        __syncthreads();
    }
    if (threadIdx.x == 0) *gidx = si[0];
}

__global__ void k_build_h0(const float* __restrict__ ge, const int* __restrict__ gidx,
                           float* __restrict__ h0, float* __restrict__ hi, int Nn)
{   size_t idx = (size_t)blockIdx.x * 256 + threadIdx.x;
    if (idx >= (size_t)Nn * EMB) return;
    int i = (int)(idx >> 7), j = (int)(idx & 127);
    float v = (i == *gidx) ? ge[j] : 0.f;
    h0[idx] = v; hi[idx] = v;
}

__global__ void k_pack_cat4(const float* __restrict__ a, const float* __restrict__ b,
                            const float* __restrict__ c, const float* __restrict__ d,
                            half_t* __restrict__ out, int M)
{   size_t idx = (size_t)blockIdx.x * 256 + threadIdx.x;
    if (idx >= (size_t)M * 512) return;
    int i = (int)(idx >> 9), j = (int)(idx & 511);
    const float* s; int jj;
    if (j < 128)      { s = a; jj = j; }
    else if (j < 256) { s = b; jj = j - 128; }
    else if (j < 384) { s = c; jj = j - 256; }
    else              { s = d; jj = j - 384; }
    out[idx] = (half_t)s[(size_t)i * EMB + jj];
}

__global__ void k_pack_cat2(const float* __restrict__ a, const float* __restrict__ b,
                            half_t* __restrict__ out, int M)
{   size_t idx = (size_t)blockIdx.x * 256 + threadIdx.x;
    if (idx >= (size_t)M * 256) return;
    int i = (int)(idx >> 8), j = (int)(idx & 255);
    float v = (j < 128) ? a[(size_t)i * EMB + j] : b[(size_t)i * EMB + (j - 128)];
    out[idx] = (half_t)v;
}

__global__ void k_pack_msg(const float* __restrict__ enc, const int* __restrict__ src,
                           const int* __restrict__ dst, half_t* __restrict__ msgin, int Ne)
{   // cols 0..383 of [Ne,640]: [xj-xi, xj, xi]
    size_t idx = (size_t)blockIdx.x * 256 + threadIdx.x;
    if (idx >= (size_t)Ne * EMB) return;
    int e = (int)(idx >> 7), j = (int)(idx & 127);
    float xj = enc[(size_t)src[e] * EMB + j];
    float xi = enc[(size_t)dst[e] * EMB + j];
    half_t* r = msgin + (size_t)e * 640;
    r[j] = (half_t)(xj - xi); r[128 + j] = (half_t)xj; r[256 + j] = (half_t)xi;
}

__global__ void k_pack_edge_attr(const half_t* __restrict__ efH, const half_t* __restrict__ ecH,
                                 half_t* __restrict__ msgin, int Ne)
{   // cols 384..639 of [Ne,640]: [ef_d, edge_code]
    size_t idx = (size_t)blockIdx.x * 256 + threadIdx.x;
    if (idx >= (size_t)Ne * EMB) return;
    int e = (int)(idx >> 7), j = (int)(idx & 127);
    half_t* r = msgin + (size_t)e * 640;
    r[384 + j] = efH[idx]; r[512 + j] = ecH[idx];
}

__global__ void k_scatter_max(const float* __restrict__ msg, const int* __restrict__ dst,
                              unsigned* __restrict__ agg, int Ne)
{   size_t idx = (size_t)blockIdx.x * 256 + threadIdx.x;
    if (idx >= (size_t)Ne * EMB) return;
    int e = (int)(idx >> 7), j = (int)(idx & 127);
    unsigned b = __float_as_uint(msg[idx]);
    unsigned key = (b & 0x80000000u) ? ~b : (b | 0x80000000u);   // monotonic float->uint
    atomicMax(&agg[(size_t)dst[e] * EMB + j], key);
}

__global__ void k_pack_enc_agg(const float* __restrict__ enc, const unsigned* __restrict__ agg,
                               half_t* __restrict__ out, int Nn)
{   size_t idx = (size_t)blockIdx.x * 256 + threadIdx.x;
    if (idx >= (size_t)Nn * 256) return;
    int i = (int)(idx >> 8), j = (int)(idx & 255);
    float vv;
    if (j < 128) vv = enc[(size_t)i * EMB + j];
    else {
        unsigned key = agg[(size_t)i * EMB + (j - 128)];
        unsigned b = (key & 0x80000000u) ? (key ^ 0x80000000u) : ~key;
        vv = __uint_as_float(b);
        if (b == 0xFF800000u) vv = 0.f;     // -inf (no in-edges) -> 0
    }
    out[idx] = (half_t)vv;
}

__global__ void k_pack_policy(const float* __restrict__ dec, const float* __restrict__ ef,
                              const int* __restrict__ src, const int* __restrict__ dst,
                              half_t* __restrict__ out, int Ne)
{   // [dec[src], dec[src]-dec[dst], ef_d] -> [Ne,384]
    size_t idx = (size_t)blockIdx.x * 256 + threadIdx.x;
    if (idx >= (size_t)Ne * EMB) return;
    int e = (int)(idx >> 7), j = (int)(idx & 127);
    float ds = dec[(size_t)src[e] * EMB + j];
    float dd = dec[(size_t)dst[e] * EMB + j];
    half_t* r = out + (size_t)e * 384;
    r[j] = (half_t)ds; r[128 + j] = (half_t)(ds - dd); r[256 + j] = (half_t)ef[idx];
}

__launch_bounds__(128)
__global__ void k_pol_dot(const float* __restrict__ x, const float* __restrict__ w,
                          float* __restrict__ pol, int Ne)
{   const int lane = threadIdx.x & 31;
    const int e = blockIdx.x * 4 + (threadIdx.x >> 5);
    if (e >= Ne) return;
    const float* xr = x + (size_t)e * EMB;
    float s = 0.f;
#pragma unroll
    for (int i = 0; i < 4; ++i) s += xr[lane + 32 * i] * w[lane + 32 * i];
#pragma unroll
    for (int d = 16; d >= 1; d >>= 1) s += __shfl_xor(s, d, 32);
    if (lane == 0) pol[e] = s;
}

__global__ void k_scatter_idx(const int* __restrict__ src, const int* __restrict__ dst,
                              unsigned* __restrict__ aux, int Ne, int Nn)
{   size_t e = (size_t)blockIdx.x * 256 + threadIdx.x;
    if (e >= (size_t)Ne) return;
    atomicMax(&aux[(size_t)dst[e] * Nn + src[e]], (unsigned)(e + 1));
}

__global__ void k_scatter_out(const float* __restrict__ pol, const int* __restrict__ src,
                              const int* __restrict__ dst, const unsigned* __restrict__ aux,
                              float* __restrict__ out, int Ne, int Nn)
{   size_t e = (size_t)blockIdx.x * 256 + threadIdx.x;
    if (e >= (size_t)Ne) return;
    size_t pos = (size_t)dst[e] * Nn + src[e];
    if (aux[pos] == (unsigned)(e + 1)) out[pos] = pol[e];   // deterministic last-wins
}

// ======================= launcher =======================
extern "C" void kernel_launch(void* const* d_in, const int* in_sizes, int n_in,
                              void* d_out, int out_size, void* d_ws, size_t ws_size,
                              hipStream_t stream)
{
    const float* goal = (const float*)d_in[0];
    const float* v    = (const float*)d_in[1];
    const float* obst = (const float*)d_in[2];
    const int*   ei   = (const int*)d_in[5];
    const int Cc = 7, Oo = 6;
    const int Nn = in_sizes[1] / Cc;
    const int No = in_sizes[2] / Oo;
    const int Ne = in_sizes[5] / 2;
    const int* src = ei;
    const int* dst = ei + Ne;
    const int LOOPN = 5;                   // reference LOOP (device scalar not readable in capture)
    const int NoPad = ((No + 127) / 128) * 128;
    const float invt = 1.0f / sqrtf((float)EMB);

    // ---- params: jax pytree flatten order (dict keys sorted alphabetically) ----
    int pi = 8;
    auto P = [&]() { return (const float*)d_in[pi++]; };
    struct LinP { const float* b; const float* w; };
    auto rdLin = [&]() { LinP l; l.b = P(); l.w = P(); return l; };
    struct BlkP {
        const float *aln_b, *aln_w, *wk, *wq, *wv;
        LinP ml1, ml2; const float *mln_b, *mln_w;
        LinP ol1, ol2; const float *oln_b, *oln_w;
    };
    auto rdBlk = [&]() {
        BlkP b;
        b.aln_b = P(); b.aln_w = P(); b.wk = P(); b.wq = P(); b.wv = P();
        b.ml1 = rdLin(); b.ml2 = rdLin(); b.mln_b = P(); b.mln_w = P();
        b.ol1 = rdLin(); b.ol2 = rdLin(); b.oln_b = P(); b.oln_w = P();
        return b;
    };
    LinP decoder = rdLin();
    BlkP eb[3]; for (int i = 0; i < 3; ++i) eb[i] = rdBlk();
    LinP ec_l1 = rdLin(), ec_l2 = rdLin();
    LinP ef_l1 = rdLin(), ef_l2 = rdLin();
    LinP encoder = rdLin();
    const float* goalEnc = P();
    LinP m0l1 = rdLin(), m0l2 = rdLin(), m1 = rdLin();
    BlkP nb[3]; for (int i = 0; i < 3; ++i) nb[i] = rdBlk();
    LinP nc_l1 = rdLin(), nc_l2 = rdLin();
    LinP nfc_l1 = rdLin(), nfc_l2 = rdLin();
    LinP oe_l1 = rdLin(), oe_l2 = rdLin();
    LinP on_l1 = rdLin(), on_l2 = rdLin();
    LinP p_l1 = rdLin(), p_l2 = rdLin();
    const float* p_l3w = P();

    // ---- workspace bump allocator (256B aligned) ----
    uintptr_t cur = (uintptr_t)d_ws;
    auto alloc = [&](size_t bytes) -> void* {
        cur = (cur + 255) & ~(uintptr_t)255;
        void* p = (void*)cur; cur += bytes; return p;
    };
    auto cdiv = [](size_t a, size_t b) -> unsigned { return (unsigned)((a + b - 1) / b); };

    auto prepW = [&](const float* W, int K) -> half_t* {
        int Kp = (K + 31) & ~31;
        half_t* Wt = (half_t*)alloc((size_t)EMB * Kp * sizeof(half_t));
        k_prep_weight<<<cdiv((size_t)EMB * Kp, 256), 256, 0, stream>>>(W, Wt, K, Kp);
        return Wt;
    };

    // prepped (transposed, padded) f16 weights
    struct BlkW { half_t *wk, *wq, *wv, *ml1, *ml2, *ol1, *ol2; };
    BlkW ebw[3], nbw[3];
    for (int i = 0; i < 3; ++i) {
        ebw[i].wk = prepW(eb[i].wk, EMB);   ebw[i].wq = prepW(eb[i].wq, EMB);
        ebw[i].wv = prepW(eb[i].wv, EMB);
        ebw[i].ml1 = prepW(eb[i].ml1.w, EMB); ebw[i].ml2 = prepW(eb[i].ml2.w, EMB);
        ebw[i].ol1 = prepW(eb[i].ol1.w, EMB); ebw[i].ol2 = prepW(eb[i].ol2.w, EMB);
        nbw[i].wk = prepW(nb[i].wk, EMB);   nbw[i].wq = prepW(nb[i].wq, EMB);
        nbw[i].wv = prepW(nb[i].wv, EMB);
        nbw[i].ml1 = prepW(nb[i].ml1.w, EMB); nbw[i].ml2 = prepW(nb[i].ml2.w, EMB);
        nbw[i].ol1 = prepW(nb[i].ol1.w, EMB); nbw[i].ol2 = prepW(nb[i].ol2.w, EMB);
    }
    half_t* nc1T = prepW(nc_l1.w, 4 * Cc);  half_t* nc2T = prepW(nc_l2.w, EMB);
    half_t* ec1T = prepW(ec_l1.w, 2 * Cc);  half_t* ec2T = prepW(ec_l2.w, EMB);
    half_t* ef1T = prepW(ef_l1.w, 2 * Cc);  half_t* ef2T = prepW(ef_l2.w, EMB);
    half_t* nf1T = prepW(nfc_l1.w, Cc);     half_t* nf2T = prepW(nfc_l2.w, EMB);
    half_t* on1T = prepW(on_l1.w, Oo);      half_t* on2T = prepW(on_l2.w, EMB);
    half_t* oe1T = prepW(oe_l1.w, Oo);      half_t* oe2T = prepW(oe_l2.w, EMB);
    half_t* encT = prepW(encoder.w, 4 * EMB);
    half_t* m0l1T = prepW(m0l1.w, 5 * EMB); half_t* m0l2T = prepW(m0l2.w, EMB);
    half_t* m1T  = prepW(m1.w, 2 * EMB);
    half_t* decT = prepW(decoder.w, 2 * EMB);
    half_t* pl1T = prepW(p_l1.w, 3 * EMB);  half_t* pl2T = prepW(p_l2.w, EMB);

    // activation buffers
    half_t* tmpH16   = (half_t*)alloc((size_t)Ne * EMB * 2);
    float*  tmpF32   = (float*) alloc((size_t)Ne * EMB * 4);
    half_t* nodein16 = (half_t*)alloc((size_t)Nn * 32 * 2);
    half_t* edgein16 = (half_t*)alloc((size_t)Ne * 32 * 2);
    half_t* obsin16  = (half_t*)alloc((size_t)No * 32 * 2);
    half_t* vin16    = (half_t*)alloc((size_t)Nn * 32 * 2);
    float*  ncF      = (float*) alloc((size_t)Nn * EMB * 4);
    half_t* ec16     = (half_t*)alloc((size_t)Ne * EMB * 2);
    float*  efF      = (float*) alloc((size_t)Ne * EMB * 4);
    half_t* efH      = (half_t*)alloc((size_t)Ne * EMB * 2);
    float*  nfF      = (float*) alloc((size_t)Nn * EMB * 4);
    half_t* nfH      = (half_t*)alloc((size_t)Nn * EMB * 2);
    float*  onF      = (float*) alloc((size_t)No * EMB * 4);
    half_t* onH      = (half_t*)alloc((size_t)No * EMB * 2);
    float*  oeF      = (float*) alloc((size_t)No * EMB * 4);
    half_t* oeH      = (half_t*)alloc((size_t)No * EMB * 2);
    float*  mkF      = (float*) alloc((size_t)Ne * EMB * 4);
    float*  mqF      = (float*) alloc((size_t)Ne * EMB * 4);
    half_t* mqH      = (half_t*)alloc((size_t)Ne * EMB * 2);
    float*  mvF      = (float*) alloc((size_t)Ne * EMB * 4);
    half_t* scores16 = (half_t*)alloc((size_t)Ne * NoPad * 2);
    float*  att0     = (float*) alloc((size_t)Ne * 4);
    float*  attvF    = (float*) alloc((size_t)Ne * EMB * 4);
    float*  mc2F     = (float*) alloc((size_t)Ne * EMB * 4);
    half_t* mc2H     = (half_t*)alloc((size_t)Ne * EMB * 2);
    half_t* okT16    = (half_t*)alloc((size_t)NoPad * EMB * 2);
    float*  ovF      = (float*) alloc((size_t)No * EMB * 4);
    half_t* ovT16    = (half_t*)alloc((size_t)EMB * NoPad * 2);
    float*  h0F      = (float*) alloc((size_t)Nn * EMB * 4);
    float*  hiF      = (float*) alloc((size_t)Nn * EMB * 4);
    half_t* encin16  = (half_t*)alloc((size_t)Nn * 4 * EMB * 2);
    float*  encF     = (float*) alloc((size_t)Nn * EMB * 4);
    unsigned* aggU   = (unsigned*)alloc((size_t)Nn * EMB * 4);
    half_t* hiin16   = (half_t*)alloc((size_t)Nn * 2 * EMB * 2);
    half_t* decin16  = (half_t*)alloc((size_t)Nn * 2 * EMB * 2);
    float*  decF     = (float*) alloc((size_t)Nn * EMB * 4);
    half_t* msgin16  = (half_t*)alloc((size_t)Ne * 640 * 2);
    float*  polF     = (float*) alloc((size_t)Ne * 4);
    int*    gidx     = (int*)   alloc(256);
    half_t*   polin16 = msgin16;            // alias: msg input dead after MPNN loop
    unsigned* aux     = (unsigned*)msgin16; // alias: dead after policy packing

    auto gemm = [&](const half_t* A, const half_t* Bt, const float* bias,
                    float* Cf, half_t* Ch, int M, int N, int Kp, int relu) {
        dim3 g(cdiv((size_t)M, 128), N / 128);
        k_gemm_wmma<<<g, 256, 0, stream>>>(A, Bt, bias, Cf, Ch, M, N, Kp, relu);
    };

    // ---- feature encoders ----
    k_pack_node_in<<<cdiv((size_t)Nn * 32, 256), 256, 0, stream>>>(v, goal, nodein16, Nn);
    k_pack_edge_in<<<cdiv((size_t)Ne * 32, 256), 256, 0, stream>>>(v, src, dst, edgein16, Ne);
    k_pack_pad<<<cdiv((size_t)No * 32, 256), 256, 0, stream>>>(obst, obsin16, No, Oo, 32);
    k_pack_pad<<<cdiv((size_t)Nn * 32, 256), 256, 0, stream>>>(v, vin16, Nn, Cc, 32);

    gemm(nodein16, nc1T, nc_l1.b, nullptr, tmpH16, Nn, EMB, 32, 1);
    gemm(tmpH16, nc2T, nc_l2.b, ncF, nullptr, Nn, EMB, EMB, 0);
    gemm(edgein16, ec1T, ec_l1.b, nullptr, tmpH16, Ne, EMB, 32, 1);
    gemm(tmpH16, ec2T, ec_l2.b, nullptr, ec16, Ne, EMB, EMB, 0);
    gemm(edgein16, ef1T, ef_l1.b, nullptr, tmpH16, Ne, EMB, 32, 1);
    gemm(tmpH16, ef2T, ef_l2.b, efF, efH, Ne, EMB, EMB, 0);
    gemm(vin16, nf1T, nfc_l1.b, nullptr, tmpH16, Nn, EMB, 32, 1);
    gemm(tmpH16, nf2T, nfc_l2.b, nfF, nfH, Nn, EMB, EMB, 0);
    gemm(obsin16, on1T, on_l1.b, nullptr, tmpH16, No, EMB, 32, 1);
    gemm(tmpH16, on2T, on_l2.b, onF, onH, No, EMB, EMB, 0);
    gemm(obsin16, oe1T, oe_l1.b, nullptr, tmpH16, No, EMB, 32, 1);
    gemm(tmpH16, oe2T, oe_l2.b, oeF, oeH, No, EMB, EMB, 0);

    // ---- transformer blocks ----
    auto applyBlock = [&](const BlkP& bp, const BlkW& bw,
                          float* mcF, half_t* mcH, int Mm, float* ocF, half_t* ocH) {
        k_fill_h16<<<cdiv((size_t)NoPad * EMB, 256), 256, 0, stream>>>(okT16, (size_t)NoPad * EMB);
        gemm(mcH, bw.wk, nullptr, mkF, nullptr, Mm, EMB, EMB, 0);
        gemm(mcH, bw.wq, nullptr, mqF, mqH, Mm, EMB, EMB, 0);
        gemm(mcH, bw.wv, nullptr, mvF, nullptr, Mm, EMB, EMB, 0);
        gemm(ocH, bw.wk, nullptr, nullptr, okT16, No, EMB, EMB, 0);
        gemm(ocH, bw.wv, nullptr, ovF, nullptr, No, EMB, EMB, 0);
        gemm(mqH, okT16, nullptr, nullptr, scores16, Mm, NoPad, EMB, 0);   // mq @ ok^T
        k_transpose_pad<<<cdiv((size_t)EMB * NoPad, 256), 256, 0, stream>>>(ovF, ovT16, No, NoPad);
        k_softmax_att<<<cdiv((size_t)Mm, 4), 128, 0, stream>>>(mqF, mkF, scores16, att0, Mm, No, NoPad, invt);
        gemm(scores16, ovT16, nullptr, attvF, nullptr, Mm, EMB, NoPad, 0); // att @ ov
        k_res_ln<<<cdiv((size_t)Mm, 4), 128, 0, stream>>>(mvF, attvF, mcF, att0,
                                                          bp.aln_w, bp.aln_b, mc2F, mc2H, Mm);
        gemm(mc2H, bw.ml1, bp.ml1.b, nullptr, tmpH16, Mm, EMB, EMB, 1);
        gemm(tmpH16, bw.ml2, bp.ml2.b, tmpF32, nullptr, Mm, EMB, EMB, 0);
        k_res_ln<<<cdiv((size_t)Mm, 4), 128, 0, stream>>>(tmpF32, mc2F, nullptr, nullptr,
                                                          bp.mln_w, bp.mln_b, mcF, mcH, Mm);
        gemm(ocH, bw.ol1, bp.ol1.b, nullptr, tmpH16, No, EMB, EMB, 1);
        gemm(tmpH16, bw.ol2, bp.ol2.b, tmpF32, nullptr, No, EMB, EMB, 0);
        k_res_ln<<<cdiv((size_t)No, 4), 128, 0, stream>>>(tmpF32, ocF, nullptr, nullptr,
                                                          bp.oln_w, bp.oln_b, ocF, ocH, No);
    };
    for (int b = 0; b < 3; ++b) {
        applyBlock(nb[b], nbw[b], nfF, nfH, Nn, onF, onH);
        applyBlock(eb[b], ebw[b], efF, efH, Ne, oeF, oeH);
    }

    // ---- MPNN ----
    k_pack_edge_attr<<<cdiv((size_t)Ne * EMB, 256), 256, 0, stream>>>(efH, ec16, msgin16, Ne);
    k_argmin_goal<<<1, 256, 0, stream>>>(v, goal, gidx, Nn);
    k_build_h0<<<cdiv((size_t)Nn * EMB, 256), 256, 0, stream>>>(goalEnc, gidx, h0F, hiF, Nn);

    for (int it = 0; it < LOOPN; ++it) {
        k_pack_cat4<<<cdiv((size_t)Nn * 512, 256), 256, 0, stream>>>(ncF, nfF, h0F, hiF, encin16, Nn);
        gemm(encin16, encT, encoder.b, encF, nullptr, Nn, EMB, 4 * EMB, 0);
        k_pack_msg<<<cdiv((size_t)Ne * EMB, 256), 256, 0, stream>>>(encF, src, dst, msgin16, Ne);
        gemm(msgin16, m0l1T, m0l1.b, nullptr, tmpH16, Ne, EMB, 5 * EMB, 1);
        gemm(tmpH16, m0l2T, m0l2.b, tmpF32, nullptr, Ne, EMB, EMB, 0);     // msg
        k_fill_u32<<<cdiv((size_t)Nn * EMB, 256), 256, 0, stream>>>(aggU, (size_t)Nn * EMB, 0x007FFFFFu);
        k_scatter_max<<<cdiv((size_t)Ne * EMB, 256), 256, 0, stream>>>(tmpF32, dst, aggU, Ne);
        k_pack_enc_agg<<<cdiv((size_t)Nn * 256, 256), 256, 0, stream>>>(encF, aggU, hiin16, Nn);
        gemm(hiin16, m1T, m1.b, hiF, nullptr, Nn, EMB, 2 * EMB, 0);
    }

    // ---- decode + policy ----
    k_pack_cat2<<<cdiv((size_t)Nn * 256, 256), 256, 0, stream>>>(ncF, hiF, decin16, Nn);
    gemm(decin16, decT, decoder.b, decF, nullptr, Nn, EMB, 2 * EMB, 0);
    k_pack_policy<<<cdiv((size_t)Ne * EMB, 256), 256, 0, stream>>>(decF, efF, src, dst, polin16, Ne);
    gemm(polin16, pl1T, p_l1.b, nullptr, tmpH16, Ne, EMB, 3 * EMB, 1);
    gemm(tmpH16, pl2T, p_l2.b, tmpF32, nullptr, Ne, EMB, EMB, 1);
    k_pol_dot<<<cdiv((size_t)Ne, 4), 128, 0, stream>>>(tmpF32, p_l3w, polF, Ne);

    // ---- scatter into [Nn,Nn] output ----
    k_fill_u32<<<cdiv((size_t)Nn * Nn, 256), 256, 0, stream>>>(aux, (size_t)Nn * Nn, 0u);
    k_scatter_idx<<<cdiv((size_t)Ne, 256), 256, 0, stream>>>(src, dst, aux, Ne, Nn);
    k_zero_f32<<<cdiv((size_t)Nn * Nn, 256), 256, 0, stream>>>((float*)d_out, (size_t)Nn * Nn);
    k_scatter_out<<<cdiv((size_t)Ne, 256), 256, 0, stream>>>(polF, src, dst, aux, (float*)d_out, Ne, Nn);

    (void)n_in; (void)out_size; (void)ws_size;
}
